// GPSLayer_24799141167762
// MI455X (gfx1250) — matrix operand order repaired and checked by reading, not compiled, not run on hardware
//
#include <hip/hip_runtime.h>
#include <hip/hip_bf16.h>
#include <math.h>
#include <stdint.h>

typedef __attribute__((ext_vector_type(16))) __bf16 v16bf;
typedef __attribute__((ext_vector_type(8)))  __bf16 v8bf;
typedef __attribute__((ext_vector_type(8)))  float  v8f;

#define NN 8192
#define EE 262144
#define DD 256
#define FF 512
#define MM 256

__device__ __forceinline__ v8f vzero8() {
  v8f z;
#pragma unroll
  for (int i = 0; i < 8; ++i) z[i] = 0.0f;
  return z;
}

// A-fragment (16x32 bf16): lane holds one contiguous 16-element run at
// k = half*16 (half = lane>=16).
__device__ __forceinline__ v16bf ld_a_frag(const __bf16* p) {
  v8bf lo = *(const v8bf*)(p);
  v8bf hi = *(const v8bf*)(p + 8);
  v16bf r;
#pragma unroll
  for (int i = 0; i < 8; ++i) { r[i] = lo[i]; r[i + 8] = hi[i]; }
  return r;
}

__device__ __forceinline__ float redmax16(float x) {
#pragma unroll
  for (int m = 1; m <= 8; m <<= 1) x = fmaxf(x, __shfl_xor(x, m, 32));
  return x;
}
__device__ __forceinline__ float redsum16(float x) {
#pragma unroll
  for (int m = 1; m <= 8; m <<= 1) x += __shfl_xor(x, m, 32);
  return x;
}

// async global->LDS copy of 16 bytes (gfx1250, ASYNCcnt-tracked)
__device__ __forceinline__ void async_cp16(unsigned lds_off, const void* gsrc) {
  asm volatile("global_load_async_to_lds_b128 %0, %1, off"
               :: "v"(lds_off), "v"((uint64_t)(uintptr_t)gsrc) : "memory");
}
__device__ __forceinline__ void wait_async0() {
#if __has_builtin(__builtin_amdgcn_s_wait_asynccnt)
  __builtin_amdgcn_s_wait_asynccnt(0);
#else
  asm volatile("s_wait_asynccnt 0x0" ::: "memory");
#endif
}

// ---------------- GCN: CSR build (counting sort by dst) + gather ----------------
__global__ void k_cnt_init(int* __restrict__ cnt) {
  int i = blockIdx.x * 256 + threadIdx.x;
  if (i < NN) cnt[i] = 0;
}
__global__ void k_cnt(const int* __restrict__ dst, int* __restrict__ cnt) {
  int e = blockIdx.x * 256 + threadIdx.x;
  if (e < EE) atomicAdd(&cnt[dst[e]], 1);
}
__global__ void k_dinv(const int* __restrict__ cnt, float* __restrict__ dinv) {
  int i = blockIdx.x * 256 + threadIdx.x;
  if (i < NN) dinv[i] = rsqrtf((float)cnt[i] + 1.0f);  // +1 self loop
}
// single-workgroup exclusive scan over NN counts -> off[NN+1], cur[NN]
__global__ __launch_bounds__(256) void k_scan(const int* __restrict__ cnt,
                                              int* __restrict__ off,
                                              int* __restrict__ cur) {
  __shared__ int part[256];
  const int t = threadIdx.x;
  const int base = t * 32;
  int loc[32];
  int s = 0;
#pragma unroll
  for (int i = 0; i < 32; ++i) { loc[i] = s; s += cnt[base + i]; }
  part[t] = s;
  __syncthreads();
  for (int d = 1; d < 256; d <<= 1) {
    int v = (t >= d) ? part[t - d] : 0;
    __syncthreads();
    part[t] += v;
    __syncthreads();
  }
  int cbase = (t == 0) ? 0 : part[t - 1];
#pragma unroll
  for (int i = 0; i < 32; ++i) {
    int o = cbase + loc[i];
    off[base + i] = o;
    cur[base + i] = o;
  }
  if (t == 255) off[NN] = part[255];
}
__global__ void k_bucket(const int* __restrict__ src, const int* __restrict__ dst,
                         int* __restrict__ cur, int* __restrict__ srcs) {
  int e = blockIdx.x * 256 + threadIdx.x;
  if (e < EE) {
    int d = dst[e];
    int p = atomicAdd(&cur[d], 1);
    srcs[p] = src[e];
  }
}
// one block per node, one thread per dim: h_local = bias + self + sum_in coef*xw[src]
__global__ __launch_bounds__(256) void k_gather(const int* __restrict__ off,
                                                const int* __restrict__ srcs,
                                                const float* __restrict__ xw,
                                                const float* __restrict__ dinv,
                                                const float* __restrict__ bgcn,
                                                float* __restrict__ hl) {
  const int i = blockIdx.x, d = threadIdx.x;
  const float di = dinv[i];
  float acc = bgcn[d] + xw[(size_t)i * DD + d] * di * di;
  const int s0 = off[i], s1 = off[i + 1];
  for (int e = s0; e < s1; ++e) {
    int s = srcs[e];
    acc += xw[(size_t)s * DD + d] * (dinv[s] * di);
  }
  hl[(size_t)i * DD + d] = acc;
}

// ---------------- fused residual + LayerNorm (D=256, one row per block) ----------------
__global__ __launch_bounds__(256) void k_add_ln(const float* __restrict__ a,
                                                const float* __restrict__ b,
                                                const float* __restrict__ g,
                                                const float* __restrict__ be,
                                                float* __restrict__ out) {
  __shared__ float red[8];
  const int row = blockIdx.x, tid = threadIdx.x;
  const int wave = tid >> 5, lane = tid & 31;
  float x = a[(size_t)row * DD + tid] + b[(size_t)row * DD + tid];
  float s = x;
#pragma unroll
  for (int m = 1; m <= 16; m <<= 1) s += __shfl_xor(s, m, 32);
  if (lane == 0) red[wave] = s;
  __syncthreads();
  float tot = 0.0f;
#pragma unroll
  for (int i = 0; i < 8; ++i) tot += red[i];
  float mu = tot * (1.0f / DD);
  float d = x - mu;
  float s2 = d * d;
#pragma unroll
  for (int m = 1; m <= 16; m <<= 1) s2 += __shfl_xor(s2, m, 32);
  __syncthreads();
  if (lane == 0) red[wave] = s2;
  __syncthreads();
  float tv = 0.0f;
#pragma unroll
  for (int i = 0; i < 8; ++i) tv += red[i];
  float var = tv * (1.0f / DD);
  out[(size_t)row * DD + tid] = d * rsqrtf(var + 1e-5f) * g[tid] + be[tid];
}

// ---------------- tiled fp32 -> bf16 transpose: dst[d][n] = src[n][d] ----------------
__global__ __launch_bounds__(256) void k_transpose_bf16(const float* __restrict__ src,
                                                        __bf16* __restrict__ dst) {
  __shared__ __bf16 t[32][33];
  const int bx = blockIdx.x;  // dim tile (DD/32)
  const int by = blockIdx.y;  // node tile (NN/32)
  const int lx = threadIdx.x & 31, ly = threadIdx.x >> 5;  // 32x8
#pragma unroll
  for (int i = 0; i < 32; i += 8)
    t[ly + i][lx] = (__bf16)src[(size_t)(by * 32 + ly + i) * DD + bx * 32 + lx];
  __syncthreads();
#pragma unroll
  for (int i = 0; i < 32; i += 8)
    dst[(size_t)(bx * 32 + ly + i) * NN + by * 32 + lx] = t[lx][ly + i];
}

// ---------------- generic bf16-WMMA GEMM (compile-time variants) ----------------
// C = A[8192,K] @ B (+bias, relu); OUTB16=0: fp32 out; OUTB16=1: bf16 out (val*scale).
// Block tile 128x64, 8 waves (4 row x 2 col), 32x32 per wave, K-step 32.
template <int TRANSB, int OUTB16, int RELU>
__global__ __launch_bounds__(256) void k_gemm_bf16(const float* __restrict__ A,
                                                   const float* __restrict__ B,
                                                   const float* __restrict__ bias,
                                                   float* __restrict__ Cf,
                                                   __bf16* __restrict__ Cb,
                                                   int K, int Ncols, float scale) {
  __shared__ __align__(32) __bf16 a_s[128 * 32];   // [row][k]
  __shared__ __align__(32) __bf16 bT_s[64 * 32];   // [col][k]
  const int tid = threadIdx.x;
  const int wave = tid >> 5, lane = tid & 31;
  const int half = lane >> 4, l16 = lane & 15;
  const int wr = wave >> 1, wc = wave & 1;
  const int row0 = blockIdx.y * 128;
  const int c0 = blockIdx.x * 64;

  v8f acc[2][2];
#pragma unroll
  for (int i = 0; i < 2; ++i)
#pragma unroll
    for (int j = 0; j < 2; ++j) acc[i][j] = vzero8();

  for (int kb = 0; kb < K; kb += 32) {
#pragma unroll
    for (int i = 0; i < 16; ++i) {  // 128*32 elems
      int idx = tid + i * 256;
      int r = idx >> 5, kk = idx & 31;
      a_s[idx] = (__bf16)A[(size_t)(row0 + r) * K + kb + kk];
    }
#pragma unroll
    for (int i = 0; i < 8; ++i) {  // 64*32 elems
      int idx = tid + i * 256;
      int n = idx >> 5, kk = idx & 31;
      float v = TRANSB ? B[(size_t)(c0 + n) * K + kb + kk]
                       : B[(size_t)(kb + kk) * Ncols + c0 + n];
      bT_s[idx] = (__bf16)v;
    }
    __syncthreads();
    v16bf af[2], bf2[2];
#pragma unroll
    for (int fr = 0; fr < 2; ++fr) {
      int arow = wr * 32 + fr * 16 + l16;
      af[fr] = ld_a_frag(&a_s[arow * 32 + half * 16]);
    }
#pragma unroll
    for (int fc = 0; fc < 2; ++fc) {
      int bcol = wc * 32 + fc * 16 + l16;
      bf2[fc] = *(const v16bf*)&bT_s[bcol * 32 + half * 16];
    }
#pragma unroll
    for (int fr = 0; fr < 2; ++fr)
#pragma unroll
      for (int fc = 0; fc < 2; ++fc)
        acc[fr][fc] = __builtin_amdgcn_wmma_f32_16x16x32_bf16(
            false, af[fr], false, bf2[fc], (short)0, acc[fr][fc], false, false);
    __syncthreads();
  }
#pragma unroll
  for (int fr = 0; fr < 2; ++fr)
#pragma unroll
    for (int fc = 0; fc < 2; ++fc) {
      int col = c0 + wc * 32 + fc * 16 + l16;
      float bb = bias ? bias[col] : 0.0f;
#pragma unroll
      for (int vv = 0; vv < 8; ++vv) {
        int row = row0 + wr * 32 + fr * 16 + vv + half * 8;
        float val = acc[fr][fc][vv] + bb;
        if (RELU) val = fmaxf(val, 0.0f);
        if (OUTB16) Cb[(size_t)row * Ncols + col] = (__bf16)(val * scale);
        else        Cf[(size_t)row * Ncols + col] = val;
      }
    }
}

// ---------------- flash attention: out = softmax(qf@kf^T / 16) @ v ----------------
// qfb: bf16 [N][M], pre-scaled by 1/16 (exact power of two). kfb: bf16 [N][M].
// vTb: bf16 [D][N] (v transposed). 64 q rows / block (4 waves x 16 rows),
// key tiles of 64, double-buffered async global->LDS staging, online softmax.
__global__ __launch_bounds__(128) void k_flash_attn(const __bf16* __restrict__ qfb,
                                                    const __bf16* __restrict__ kfb,
                                                    const __bf16* __restrict__ vTb,
                                                    float* __restrict__ out) {
  __shared__ __align__(32) __bf16 kf_s[2][64 * 256];   // [key][dim]  2x32KB
  __shared__ __align__(32) __bf16 vT_s[2][256 * 64];   // [dim][key]  2x32KB
  __shared__ __align__(32) __bf16 p_s[4 * 16 * 64];    // per-wave P scratch 8KB
  const int tid = threadIdx.x;
  const int wave = tid >> 5, lane = tid & 31;
  const int half = lane >> 4, l16 = lane & 15;
  const int row0 = blockIdx.x * 64;

  const unsigned kf_base = (unsigned)(uintptr_t)(void*)&kf_s[0][0];
  const unsigned vT_base = (unsigned)(uintptr_t)(void*)&vT_s[0][0];

  auto stage = [&](int j, int buf) {
    const int k0 = j * 64;
    const unsigned kb = kf_base + (unsigned)buf * 32768u;
    const unsigned vb = vT_base + (unsigned)buf * 32768u;
    const char* kg = (const char*)(kfb + (size_t)k0 * MM);
#pragma unroll
    for (int i = 0; i < 16; ++i) {
      int c = tid + i * 128;  // 2048 x 16B chunks (32KB contiguous)
      async_cp16(kb + c * 16, kg + c * 16);
    }
#pragma unroll
    for (int i = 0; i < 16; ++i) {
      int c = tid + i * 128;
      int d = c >> 2, q4 = c & 3;  // row d (128B), 16B sub-chunk q4
      async_cp16(vb + d * 128 + q4 * 16,
                 (const char*)(vTb + (size_t)d * NN + k0) + q4 * 16);
    }
  };

  // qf A-fragments for this wave's 16 rows, whole K=256, kept in registers
  v16bf aq[8];
  {
    const __bf16* qp = qfb + (size_t)(row0 + wave * 16 + l16) * MM;
#pragma unroll
    for (int kc = 0; kc < 8; ++kc)
      aq[kc] = ld_a_frag(qp + kc * 32 + half * 16);
  }

  v8f accO[16];
#pragma unroll
  for (int i = 0; i < 16; ++i) accO[i] = vzero8();
  float mrow[8], lrow[8];
#pragma unroll
  for (int i = 0; i < 8; ++i) { mrow[i] = -3.0e38f; lrow[i] = 0.0f; }

  __bf16* pw = &p_s[wave * 1024];

  stage(0, 0);
  for (int j = 0; j < NN / 64; ++j) {
    const int buf = j & 1;
    wait_async0();       // tile j resident (this wave's copies)
    __syncthreads();     // all waves' copies visible; prior compute on `buf^1` done
    if (j + 1 < NN / 64) stage(j + 1, buf ^ 1);  // prefetch next tile

    const __bf16* kfp = &kf_s[buf][0];
    const __bf16* vTp = &vT_s[buf][0];

    // S (16x64) = qf_tile @ kf_tile^T
    v8f S[4];
#pragma unroll
    for (int nf = 0; nf < 4; ++nf) S[nf] = vzero8();
#pragma unroll
    for (int kc = 0; kc < 8; ++kc)
#pragma unroll
      for (int nf = 0; nf < 4; ++nf) {
        int key = nf * 16 + l16;
        v16bf b = *(const v16bf*)&kfp[key * 256 + kc * 32 + half * 16];
        S[nf] = __builtin_amdgcn_wmma_f32_16x16x32_bf16(
            false, aq[kc], false, b, (short)0, S[nf], false, false);
      }

    // online softmax (rows split across wave halves exactly as C-fragment layout)
    float mnew[8], alpha[8];
#pragma unroll
    for (int r = 0; r < 8; ++r) {
      float t = S[0][r];
#pragma unroll
      for (int nf = 1; nf < 4; ++nf) t = fmaxf(t, S[nf][r]);
      t = redmax16(t);
      mnew[r] = fmaxf(mrow[r], t);
      alpha[r] = __expf(mrow[r] - mnew[r]);
    }
#pragma unroll
    for (int nf = 0; nf < 4; ++nf)
#pragma unroll
      for (int r = 0; r < 8; ++r) S[nf][r] = __expf(S[nf][r] - mnew[r]);
#pragma unroll
    for (int r = 0; r < 8; ++r) {
      float t = S[0][r] + S[1][r] + S[2][r] + S[3][r];
      t = redsum16(t);
      lrow[r] = lrow[r] * alpha[r] + t;
      mrow[r] = mnew[r];
    }
#pragma unroll
    for (int nf = 0; nf < 16; ++nf)
#pragma unroll
      for (int r = 0; r < 8; ++r) accO[nf][r] *= alpha[r];

    // P -> per-wave LDS (re-layout C-fragment -> A-fragment)
#pragma unroll
    for (int nf = 0; nf < 4; ++nf)
#pragma unroll
      for (int r = 0; r < 8; ++r)
        pw[(r + half * 8) * 64 + nf * 16 + l16] = (__bf16)S[nf][r];
    v16bf pa[2];
#pragma unroll
    for (int kc2 = 0; kc2 < 2; ++kc2)
      pa[kc2] = ld_a_frag(&pw[l16 * 64 + kc2 * 32 + half * 16]);

    // O += P (16x64) @ V (64x256)
#pragma unroll
    for (int kc2 = 0; kc2 < 2; ++kc2)
#pragma unroll
      for (int nf = 0; nf < 16; ++nf) {
        int d = nf * 16 + l16;
        v16bf b = *(const v16bf*)&vTp[d * 64 + kc2 * 32 + half * 16];
        accO[nf] = __builtin_amdgcn_wmma_f32_16x16x32_bf16(
            false, pa[kc2], false, b, (short)0, accO[nf], false, false);
      }
  }

  float inv[8];
#pragma unroll
  for (int r = 0; r < 8; ++r) inv[r] = 1.0f / lrow[r];
#pragma unroll
  for (int nf = 0; nf < 16; ++nf)
#pragma unroll
    for (int r = 0; r < 8; ++r) {
      int row = row0 + wave * 16 + r + half * 8;
      int col = nf * 16 + l16;
      out[(size_t)row * DD + col] = accO[nf][r] * inv[r];
    }
}

// ---------------- launcher ----------------
extern "C" void kernel_launch(void* const* d_in, const int* in_sizes, int n_in,
                              void* d_out, int out_size, void* d_ws, size_t ws_size,
                              hipStream_t stream) {
  (void)in_sizes; (void)n_in; (void)out_size; (void)ws_size;
  const float* h    = (const float*)d_in[0];
  const int*   ei   = (const int*)d_in[1];
  const float* Wgcn = (const float*)d_in[2];
  const float* bgcn = (const float*)d_in[3];
  const float* Wq   = (const float*)d_in[4];
  const float* bq   = (const float*)d_in[5];
  const float* Wk   = (const float*)d_in[6];
  const float* bk   = (const float*)d_in[7];
  const float* Wv   = (const float*)d_in[8];
  const float* bv   = (const float*)d_in[9];
  const float* Wo   = (const float*)d_in[10];
  const float* bo   = (const float*)d_in[11];
  const float* RF   = (const float*)d_in[12];
  const float* g1   = (const float*)d_in[13];
  const float* be1  = (const float*)d_in[14];
  const float* g2   = (const float*)d_in[15];
  const float* be2  = (const float*)d_in[16];
  const float* g3   = (const float*)d_in[17];
  const float* be3  = (const float*)d_in[18];
  const float* W1   = (const float*)d_in[19];
  const float* b1   = (const float*)d_in[20];
  const float* W2   = (const float*)d_in[21];
  const float* b2   = (const float*)d_in[22];
  float* out = (float*)d_out;

  float* ws = (float*)d_ws;
  const size_t SLOT = (size_t)NN * DD;  // 2M floats = 8 MB
  float* dinv = ws;                     // NN floats
  float* bufA = ws + 8192;
  float* bufB = bufA + SLOT;
  float* bufC = bufB + SLOT;
  float* bufD = bufC + SLOT;
  float* bufE = bufD + SLOT;
  float* bufF = bufE + SLOT;            // NN*FF floats = 16 MB
  int*   cnt  = (int*)(bufF + (size_t)NN * FF);
  int*   off  = cnt + NN;               // NN+1
  int*   cur  = off + NN + 1;
  int*   srcs = cur + NN;               // EE

  __bf16* qf16 = (__bf16*)bufE;         // [NN][MM] bf16 (4MB of slot E)
  __bf16* kf16 = (__bf16*)bufA;         // [NN][MM] bf16 (slot A, xw dead)
  __bf16* vT16 = (__bf16*)bufD;         // [DD][NN] bf16 (slot D, k dead)

  dim3 blk(256);
  dim3 g256(DD / 64, NN / 128);
  dim3 g512(FF / 64, NN / 128);

  // ---- GCN local branch: CSR build + WMMA GEMM + gather ----
  k_cnt_init<<<NN / 256, blk, 0, stream>>>(cnt);
  k_cnt<<<EE / 256, blk, 0, stream>>>(ei + EE, cnt);
  k_dinv<<<NN / 256, blk, 0, stream>>>(cnt, dinv);
  k_scan<<<1, blk, 0, stream>>>(cnt, off, cur);
  k_bucket<<<EE / 256, blk, 0, stream>>>(ei, ei + EE, cur, srcs);
  k_gemm_bf16<0, 0, 0><<<g256, blk, 0, stream>>>(h, Wgcn, nullptr, bufA, nullptr, DD, DD, 1.0f); // xw
  k_gather<<<NN, blk, 0, stream>>>(off, srcs, bufA, dinv, bgcn, bufB);   // h_local
  k_add_ln<<<NN, blk, 0, stream>>>(h, bufB, g1, be1, bufC);              // h1

  // ---- global (Performer-dense) attention ----
  k_gemm_bf16<0, 0, 0><<<g256, blk, 0, stream>>>(bufC, Wq, bq, bufD, nullptr, DD, DD, 1.0f);    // q
  k_gemm_bf16<1, 1, 0><<<g256, blk, 0, stream>>>(bufD, RF, nullptr, nullptr, qf16, DD, MM, 0.0625f); // qf/16 bf16
  k_gemm_bf16<0, 0, 0><<<g256, blk, 0, stream>>>(bufC, Wk, bk, bufD, nullptr, DD, DD, 1.0f);    // k
  k_gemm_bf16<1, 1, 0><<<g256, blk, 0, stream>>>(bufD, RF, nullptr, nullptr, kf16, DD, MM, 1.0f);    // kf bf16
  k_gemm_bf16<0, 0, 0><<<g256, blk, 0, stream>>>(bufC, Wv, bv, bufB, nullptr, DD, DD, 1.0f);    // v
  k_transpose_bf16<<<dim3(DD / 32, NN / 32), blk, 0, stream>>>(bufB, vT16);                     // v^T bf16
  k_flash_attn<<<NN / 64, dim3(128), 0, stream>>>(qf16, kf16, vT16, bufB);                      // attn@v
  k_gemm_bf16<0, 0, 0><<<g256, blk, 0, stream>>>(bufB, Wo, bo, bufD, nullptr, DD, DD, 1.0f);    // h_global
  k_add_ln<<<NN, blk, 0, stream>>>(bufC, bufD, g2, be2, bufA);                                  // h2

  // ---- FFN ----
  k_gemm_bf16<0, 0, 1><<<g512, blk, 0, stream>>>(bufA, W1, b1, bufF, nullptr, DD, FF, 1.0f);    // relu(h2@W1+b1)
  k_gemm_bf16<0, 0, 0><<<g256, blk, 0, stream>>>(bufF, W2, b2, bufB, nullptr, FF, DD, 1.0f);    // @W2+b2
  k_add_ln<<<NN, blk, 0, stream>>>(bufA, bufB, g3, be3, out);                                   // final LN
}
